// SwinTransformerBlock_19344532701891
// MI455X (gfx1250) — matrix operand order repaired and checked by reading, not compiled, not run on hardware
//
#include <hip/hip_runtime.h>
#include <hip/hip_bf16.h>
#include <math.h>

// ---------------------------------------------------------------------------
// Swin block on MI455X (gfx1250, wave32).
// GEMMs: v_wmma_f32_16x16x32_f16;  A-panels staged to LDS via TDM
// (tensor_load_to_lds + s_wait_tensorcnt) with ping-pong double buffering.
// ---------------------------------------------------------------------------

typedef __attribute__((ext_vector_type(16))) _Float16 v16h;
typedef __attribute__((ext_vector_type(8)))  float    v8f;
typedef __attribute__((ext_vector_type(4)))  unsigned int v4u;
typedef __attribute__((ext_vector_type(8)))  int      v8i_;
typedef __attribute__((ext_vector_type(4)))  int      v4i_;

#define WS_   7
#define SHIFT_ 3
#define NH_   16
#define DIM_  512
#define HD_   32
#define HH_   112
#define WW_   112
#define NT_   12544
#define NWIN_ 256
#define NTOK_ 49
#define NPAD_ 64
#define HID_  2048
#define EPS_  1e-5f
#define QSCALE_ 0.17677669529663687f
#define CHUNK_ 256       // K-chunk staged into LDS per TDM op (32x256 f16 = 16KB)

// ---- workspace layout (bytes) ----
#define OFF_WQKV  ((size_t)0)
#define OFF_WPROJ ((size_t)1572864)
#define OFF_WFC1  ((size_t)2097152)
#define OFF_WFC2  ((size_t)4194304)
#define OFF_XW    ((size_t)6291456)
#define OFF_Q     ((size_t)19136512)
#define OFF_K     ((size_t)35913728)
#define OFF_V     ((size_t)52690944)
#define OFF_ATTN  ((size_t)69468160)
#define OFF_X1    ((size_t)82313216)
#define OFF_HID   OFF_Q

// ---- TDM availability ----
#if defined(__has_builtin)
#if __has_builtin(__builtin_amdgcn_tensor_load_to_lds) && \
    __has_builtin(__builtin_amdgcn_s_wait_tensorcnt)
#define HAVE_TDM 1
#endif
#endif
#ifndef HAVE_TDM
#define HAVE_TDM 0
#endif

// low 32 bits of a generic LDS pointer = wave-relative LDS byte offset
__device__ inline unsigned lds_off32(const void* p) {
  return (unsigned)(uintptr_t)p;
}

#if HAVE_TDM
// 2-D TDM load: rows x width f16 tile (row stride = stride elems) -> LDS.
// D# encoding per CDNA5 ISA ch.8: group0 {count, lds_addr, global_addr, type=2},
// group1 {data_size=2B, tensor_dim0/1, tile_dim0/1, tensor_dim0_stride}.
__device__ inline void tdm_load_tile_2d(const void* gsrc, unsigned lds_byte,
                                        unsigned width, unsigned rows,
                                        unsigned stride_elems) {
  unsigned long long ga = (unsigned long long)(uintptr_t)gsrc;
  v4u g0;
  g0[0] = 1u;                                            // count = 1 descriptor
  g0[1] = lds_byte;                                      // lds_addr (bytes)
  g0[2] = (unsigned)(ga & 0xffffffffull);                // global_addr[31:0]
  g0[3] = (unsigned)((ga >> 32) & 0x1ffffffull)          // global_addr[56:32]
          | 0x80000000u;                                 // type = 2 ("image")
  v8i_ g1;
  g1[0] = (int)(1u << 16);                               // data_size = 2 bytes
  g1[1] = (int)((width & 0xffffu) << 16);                // tensor_dim0[15:0]
  g1[2] = (int)(((width >> 16) & 0xffffu)                // tensor_dim0[31:16]
                | ((rows & 0xffffu) << 16));             // tensor_dim1[15:0]
  g1[3] = (int)(((rows >> 16) & 0xffffu)                 // tensor_dim1[31:16]
                | ((width & 0xffffu) << 16));            // tile_dim0 = width
  g1[4] = (int)(rows & 0xffffu);                         // tile_dim1 = rows
  g1[5] = (int)stride_elems;                             // tensor_dim0_stride lo
  g1[6] = 0;
  g1[7] = 0;
  v4i_ z4 = {0, 0, 0, 0};
#if __clang_major__ >= 23
  v8i_ z8 = {0, 0, 0, 0, 0, 0, 0, 0};
  __builtin_amdgcn_tensor_load_to_lds(g0, g1, z4, z4, z8, 0);
#else
  __builtin_amdgcn_tensor_load_to_lds(g0, g1, z4, z4, 0);
#endif
}
#endif

// ---------------------------------------------------------------------------
// WMMA fragment helpers (wave32; A and B both use contiguous k = 16*(lane/16) + i)
// ---------------------------------------------------------------------------
__device__ inline v16h load_a_frag(const _Float16* A, int lda) {
  const int lane = threadIdx.x & 31;
  const int hi   = lane >> 4;
  const _Float16* p = A + (size_t)(lane & 15) * lda;
  v16h a;
#pragma unroll
  for (int i = 0; i < 16; ++i) a[i] = p[hi * 16 + i];
  return a;
}

// B from W[N,K] row-major (A @ W^T): lane = column, 16 contiguous K halves.
__device__ inline v16h load_b_frag_wt(const _Float16* W, int ldw) {
  const int lane = threadIdx.x & 31;
  const _Float16* p = W + (size_t)(lane & 15) * ldw + (lane >> 4) * 16;
  v16h b;
#pragma unroll
  for (int i = 0; i < 16; ++i) b[i] = p[i];
  return b;
}

// B from row-major [K,N] matrix (P@V): B[k][n] = M[k][n].
__device__ inline v16h load_b_frag_km(const _Float16* M, int ldb) {
  const int lane = threadIdx.x & 31;
  const int col  = lane & 15;
  const int kb   = (lane >> 4) * 16;
  v16h b;
#pragma unroll
  for (int i = 0; i < 16; ++i) b[i] = M[(size_t)(kb + i) * ldb + col];
  return b;
}

__device__ inline v8f wmma_f16(v16h a, v16h b, v8f c) {
  return __builtin_amdgcn_wmma_f32_16x16x32_f16(false, a, false, b, (short)0, c,
                                                false, false);
}

__device__ inline float wave_sum32(float v) {
#pragma unroll
  for (int off = 16; off > 0; off >>= 1) v += __shfl_xor(v, off, 32);
  return v;
}

// ---------------------------------------------------------------------------
// Block-cooperative GEMM core: block = 32 rows x 256 cols, 8 waves (2m x 4n).
// A panel (32 x K) is staged through LDS in 32x256 chunks:
//   - TDM path: wave0 issues tensor_load_to_lds, double-buffered, overlapped
//   - fallback: cooperative b128 copy
// Each wave accumulates a 16x64 tile (4 x v8f).
// ---------------------------------------------------------------------------
__device__ inline void block_gemm_acc(const _Float16* __restrict__ Ablk, int K,
                                      const _Float16* __restrict__ Wn0,
                                      v8f acc[4]) {
  __shared__ _Float16 Abuf[2][32 * CHUNK_];
  const int wave = threadIdx.x >> 5;
  const int mloc = (wave >> 2) * 16;
  const int nchunk = K / CHUNK_;

#if HAVE_TDM
  if (wave == 0)
    tdm_load_tile_2d(Ablk, lds_off32(&Abuf[0][0]), CHUNK_, 32, (unsigned)K);
#endif
  for (int c = 0; c < nchunk; ++c) {
#if HAVE_TDM
    if (wave == 0) __builtin_amdgcn_s_wait_tensorcnt(0);   // chunk c landed
    __syncthreads();            // all waves: chunk c visible, chunk c-1 reads done
    if (wave == 0 && c + 1 < nchunk)                       // prefetch overlaps WMMA
      tdm_load_tile_2d(Ablk + (size_t)(c + 1) * CHUNK_,
                       lds_off32(&Abuf[(c + 1) & 1][0]), CHUNK_, 32, (unsigned)K);
    const _Float16* Al = &Abuf[c & 1][0] + (size_t)mloc * CHUNK_;
#else
    __syncthreads();
    {
      const _Float16* src = Ablk + (size_t)c * CHUNK_;
      for (int j = threadIdx.x; j < 32 * CHUNK_ / 8; j += 256) {
        int r = j >> 5;                 // 32 vec8 per row
        int o = (j & 31) * 8;
        *(uint4*)&Abuf[0][r * CHUNK_ + o] = *(const uint4*)&src[(size_t)r * K + o];
      }
    }
    __syncthreads();
    const _Float16* Al = &Abuf[0][0] + (size_t)mloc * CHUNK_;
#endif
    const _Float16* Wk = Wn0 + (size_t)c * CHUNK_;
#pragma unroll
    for (int kk = 0; kk < CHUNK_; kk += 32) {
      v16h a = load_a_frag(Al + kk, CHUNK_);
#pragma unroll
      for (int t = 0; t < 4; ++t) {
        v16h b = load_b_frag_wt(Wk + (size_t)(t * 16) * K + kk, K);
        acc[t] = wmma_f16(a, b, acc[t]);
      }
    }
  }
}

// ---------------------------------------------------------------------------
// 0) fp32 -> fp16 weight conversion
// ---------------------------------------------------------------------------
__global__ void k_cvt_f16(const float* __restrict__ src, _Float16* __restrict__ dst, int n) {
  int i = blockIdx.x * blockDim.x + threadIdx.x;
  if (i < n) dst[i] = (_Float16)src[i];
}

// ---------------------------------------------------------------------------
// 1) LayerNorm1 + cyclic shift + window partition -> xw_f16[12544,512]
// ---------------------------------------------------------------------------
__global__ void k_ln_shift_win(const float* __restrict__ x,
                               const float* __restrict__ w,
                               const float* __restrict__ b,
                               _Float16* __restrict__ xw) {
  const int wave = threadIdx.x >> 5;
  const int lane = threadIdx.x & 31;
  const int t    = blockIdx.x * 8 + wave;
  const int h = t / WW_, wcol = t % WW_;

  float v[16];
  float s = 0.f, s2 = 0.f;
#pragma unroll
  for (int i = 0; i < 16; ++i) {
    int c = i * 32 + lane;
    float f = x[(size_t)c * NT_ + t];
    v[i] = f; s += f; s2 += f * f;
  }
  s  = wave_sum32(s);
  s2 = wave_sum32(s2);
  const float mean = s * (1.0f / DIM_);
  const float inv  = rsqrtf(s2 * (1.0f / DIM_) - mean * mean + EPS_);

  const int hs = (h + HH_ - SHIFT_) % HH_;
  const int wsd = (wcol + WW_ - SHIFT_) % WW_;
  const int a = hs / WS_, i0 = hs % WS_;
  const int bb = wsd / WS_, j0 = wsd % WS_;
  const int tsw = (a * 16 + bb) * NTOK_ + i0 * WS_ + j0;
  _Float16* dst = xw + (size_t)tsw * DIM_;
#pragma unroll
  for (int i = 0; i < 16; ++i) {
    int c = i * 32 + lane;
    dst[c] = (_Float16)((v[i] - mean) * inv * w[c] + b[c]);
  }
}

// ---------------------------------------------------------------------------
// 2) QKV GEMM -> scatter to padded q/k/v.  grid (6, 392), block 256
// ---------------------------------------------------------------------------
__global__ void k_gemm_qkv(const _Float16* __restrict__ A,
                           const _Float16* __restrict__ W,
                           const float* __restrict__ bias,
                           _Float16* __restrict__ q,
                           _Float16* __restrict__ kk,
                           _Float16* __restrict__ vv) {
  const int wave = threadIdx.x >> 5;
  const int lane = threadIdx.x & 31;
  const int m0 = blockIdx.y * 32 + (wave >> 2) * 16;
  const int n0 = blockIdx.x * 256 + (wave & 3) * 64;

  v8f acc[4] = {v8f{}, v8f{}, v8f{}, v8f{}};
  block_gemm_acc(A + (size_t)blockIdx.y * 32 * DIM_, DIM_,
                 W + (size_t)n0 * DIM_, acc);

  const int hi = lane >> 4;
#pragma unroll
  for (int tle = 0; tle < 4; ++tle) {
    const int col = n0 + tle * 16 + (lane & 15);
    const int sec  = col >> 9;
    const int cc   = col & 511;
    const int head = cc >> 5, d = cc & 31;
    const float bcol = bias[col];
#pragma unroll
    for (int j = 0; j < 8; ++j) {
      const int tok = m0 + j + hi * 8;
      const int win = tok / NTOK_, n = tok - win * NTOK_;
      float val = acc[tle][j] + bcol;
      const size_t idx = (((size_t)win * NH_ + head) * NPAD_ + n) * HD_ + d;
      if (sec == 0)      q [idx] = (_Float16)(val * QSCALE_);
      else if (sec == 1) kk[idx] = (_Float16)val;
      else               vv[idx] = (_Float16)val;
    }
  }
}

// ---------------------------------------------------------------------------
// 3) Attention: one wave per (window, head).
// ---------------------------------------------------------------------------
__global__ __launch_bounds__(32)
void k_attn(const _Float16* __restrict__ q,
            const _Float16* __restrict__ kmat,
            const _Float16* __restrict__ vmat,
            const float* __restrict__ rpb,
            _Float16* __restrict__ out) {
  __shared__ float    S[NPAD_ * NPAD_];
  __shared__ _Float16 P[NPAD_ * NPAD_];
  __shared__ int      label[NPAD_];

  const int wh   = blockIdx.x;
  const int win  = wh >> 4;
  const int head = wh & 15;
  const int lane = threadIdx.x;
  const int hi   = lane >> 4;
  const int a = win >> 4, bwin = win & 15;

  const _Float16* Q = q    + (size_t)wh * NPAD_ * HD_;
  const _Float16* K = kmat + (size_t)wh * NPAD_ * HD_;
  const _Float16* V = vmat + (size_t)wh * NPAD_ * HD_;

  for (int n = lane; n < NPAD_; n += 32) {
    if (n < NTOK_) {
      int hs = a * WS_ + n / WS_;
      int ws = bwin * WS_ + n % WS_;
      int rh = (hs < HH_ - WS_) ? 0 : ((hs < HH_ - SHIFT_) ? 1 : 2);
      int rw = (ws < WW_ - WS_) ? 0 : ((ws < WW_ - SHIFT_) ? 1 : 2);
      label[n] = rh * 3 + rw;
    } else label[n] = -1;
  }
  for (int i = lane; i < NPAD_ * NPAD_; i += 32) P[i] = (_Float16)0.f;
  __syncthreads();

  // S = Q @ K^T (q pre-scaled)
#pragma unroll
  for (int mt = 0; mt < 4; ++mt) {
    v16h af = load_a_frag(Q + (size_t)(mt * 16) * HD_, HD_);
#pragma unroll
    for (int nt = 0; nt < 4; ++nt) {
      v16h bf = load_b_frag_wt(K + (size_t)(nt * 16) * HD_, HD_);
      v8f c = {};
      c = wmma_f16(af, bf, c);
#pragma unroll
      for (int j = 0; j < 8; ++j)
        S[(mt * 16 + j + hi * 8) * NPAD_ + nt * 16 + (lane & 15)] = c[j];
    }
  }
  __syncthreads();

  // bias + mask + softmax over 49 columns
  for (int r = lane; r < NTOK_; r += 32) {
    const int i1 = r / WS_, j1 = r % WS_;
    const int lr = label[r];
    float mx = -1e30f;
    for (int c = 0; c < NTOK_; ++c) {
      const int i2 = c / WS_, j2 = c % WS_;
      const int rel = (i1 - i2 + WS_ - 1) * (2 * WS_ - 1) + (j1 - j2 + WS_ - 1);
      float val = S[r * NPAD_ + c] + rpb[rel * NH_ + head];
      if (label[c] != lr) val -= 1e9f;
      S[r * NPAD_ + c] = val;
      mx = fmaxf(mx, val);
    }
    float sum = 0.f;
    for (int c = 0; c < NTOK_; ++c) {
      float e = __expf(S[r * NPAD_ + c] - mx);
      S[r * NPAD_ + c] = e;
      sum += e;
    }
    const float inv = 1.0f / sum;
    for (int c = 0; c < NTOK_; ++c)
      P[r * NPAD_ + c] = (_Float16)(S[r * NPAD_ + c] * inv);
  }
  __syncthreads();

  // O = P @ V (A operand streamed from LDS)
#pragma unroll
  for (int mt = 0; mt < 4; ++mt) {
    v8f acc[2] = {v8f{}, v8f{}};
#pragma unroll
    for (int k0 = 0; k0 < NPAD_; k0 += 32) {
      v16h af = load_a_frag(P + (size_t)(mt * 16) * NPAD_ + k0, NPAD_);
#pragma unroll
      for (int nt = 0; nt < 2; ++nt) {
        v16h bf = load_b_frag_km(V + (size_t)k0 * HD_ + nt * 16, HD_);
        acc[nt] = wmma_f16(af, bf, acc[nt]);
      }
    }
#pragma unroll
    for (int nt = 0; nt < 2; ++nt) {
      const int d = nt * 16 + (lane & 15);
#pragma unroll
      for (int j = 0; j < 8; ++j) {
        const int m = mt * 16 + j + hi * 8;
        if (m < NTOK_)
          out[((size_t)win * NTOK_ + m) * DIM_ + head * HD_ + d] =
              (_Float16)acc[nt][j];
      }
    }
  }
}

// ---------------------------------------------------------------------------
// 4) proj GEMM + reverse shift/unpartition + residual -> x1.  grid (2, 392)
// ---------------------------------------------------------------------------
__global__ void k_gemm_proj(const _Float16* __restrict__ A,
                            const _Float16* __restrict__ W,
                            const float* __restrict__ bias,
                            const float* __restrict__ x,
                            float* __restrict__ x1) {
  const int wave = threadIdx.x >> 5;
  const int lane = threadIdx.x & 31;
  const int m0 = blockIdx.y * 32 + (wave >> 2) * 16;
  const int n0 = blockIdx.x * 256 + (wave & 3) * 64;

  v8f acc[4] = {v8f{}, v8f{}, v8f{}, v8f{}};
  block_gemm_acc(A + (size_t)blockIdx.y * 32 * DIM_, DIM_,
                 W + (size_t)n0 * DIM_, acc);

  const int hi = lane >> 4;
#pragma unroll
  for (int tle = 0; tle < 4; ++tle) {
    const int col = n0 + tle * 16 + (lane & 15);
    const float bcol = bias[col];
#pragma unroll
    for (int j = 0; j < 8; ++j) {
      const int tsw = m0 + j + hi * 8;
      const int win = tsw / NTOK_, n = tsw - win * NTOK_;
      const int hs = (win >> 4) * WS_ + n / WS_;
      const int ws = (win & 15) * WS_ + n % WS_;
      const int h = (hs + SHIFT_) % HH_;
      const int w = (ws + SHIFT_) % WW_;
      const int pos = h * WW_ + w;
      x1[(size_t)pos * DIM_ + col] =
          x[(size_t)col * NT_ + pos] + acc[tle][j] + bcol;
    }
  }
}

// ---------------------------------------------------------------------------
// 5) LayerNorm2 -> x2 f16
// ---------------------------------------------------------------------------
__global__ void k_ln2(const float* __restrict__ x1,
                      const float* __restrict__ w,
                      const float* __restrict__ b,
                      _Float16* __restrict__ x2) {
  const int wave = threadIdx.x >> 5;
  const int lane = threadIdx.x & 31;
  const int t = blockIdx.x * 8 + wave;
  const float* row = x1 + (size_t)t * DIM_;

  float v[16];
  float s = 0.f, s2 = 0.f;
#pragma unroll
  for (int i = 0; i < 16; ++i) {
    int c = i * 32 + lane;
    float f = row[c];
    v[i] = f; s += f; s2 += f * f;
  }
  s  = wave_sum32(s);
  s2 = wave_sum32(s2);
  const float mean = s * (1.0f / DIM_);
  const float inv  = rsqrtf(s2 * (1.0f / DIM_) - mean * mean + EPS_);
  _Float16* dst = x2 + (size_t)t * DIM_;
#pragma unroll
  for (int i = 0; i < 16; ++i) {
    int c = i * 32 + lane;
    dst[c] = (_Float16)((v[i] - mean) * inv * w[c] + b[c]);
  }
}

// ---------------------------------------------------------------------------
// 6) fc1 GEMM + GELU(tanh) -> hidden f16.  grid (8, 392)
// ---------------------------------------------------------------------------
__global__ void k_gemm_fc1(const _Float16* __restrict__ A,
                           const _Float16* __restrict__ W,
                           const float* __restrict__ bias,
                           _Float16* __restrict__ H) {
  const int wave = threadIdx.x >> 5;
  const int lane = threadIdx.x & 31;
  const int m0 = blockIdx.y * 32 + (wave >> 2) * 16;
  const int n0 = blockIdx.x * 256 + (wave & 3) * 64;

  v8f acc[4] = {v8f{}, v8f{}, v8f{}, v8f{}};
  block_gemm_acc(A + (size_t)blockIdx.y * 32 * DIM_, DIM_,
                 W + (size_t)n0 * DIM_, acc);

  const int hi = lane >> 4;
#pragma unroll
  for (int tle = 0; tle < 4; ++tle) {
    const int col = n0 + tle * 16 + (lane & 15);
    const float bcol = bias[col];
#pragma unroll
    for (int j = 0; j < 8; ++j) {
      const int m = m0 + j + hi * 8;
      float g = acc[tle][j] + bcol;
      g = 0.5f * g * (1.0f + tanhf(0.7978845608028654f * (g + 0.044715f * g * g * g)));
      H[(size_t)m * HID_ + col] = (_Float16)g;
    }
  }
}

// ---------------------------------------------------------------------------
// 7) fc2 GEMM + residual -> d_out [C,H,W] f32.  grid (2, 392), K=2048
// ---------------------------------------------------------------------------
__global__ void k_gemm_fc2(const _Float16* __restrict__ A,
                           const _Float16* __restrict__ W,
                           const float* __restrict__ bias,
                           const float* __restrict__ x1,
                           float* __restrict__ out) {
  const int wave = threadIdx.x >> 5;
  const int lane = threadIdx.x & 31;
  const int m0 = blockIdx.y * 32 + (wave >> 2) * 16;
  const int n0 = blockIdx.x * 256 + (wave & 3) * 64;

  v8f acc[4] = {v8f{}, v8f{}, v8f{}, v8f{}};
  block_gemm_acc(A + (size_t)blockIdx.y * 32 * HID_, HID_,
                 W + (size_t)n0 * HID_, acc);

  const int hi = lane >> 4;
#pragma unroll
  for (int tle = 0; tle < 4; ++tle) {
    const int col = n0 + tle * 16 + (lane & 15);
    const float bcol = bias[col];
#pragma unroll
    for (int j = 0; j < 8; ++j) {
      const int t = m0 + j + hi * 8;
      out[(size_t)col * NT_ + t] =
          x1[(size_t)t * DIM_ + col] + acc[tle][j] + bcol;
    }
  }
}

// ---------------------------------------------------------------------------
extern "C" void kernel_launch(void* const* d_in, const int* in_sizes, int n_in,
                              void* d_out, int out_size, void* d_ws, size_t ws_size,
                              hipStream_t stream) {
  const float* x       = (const float*)d_in[0];
  const float* norm1_w = (const float*)d_in[1];
  const float* norm1_b = (const float*)d_in[2];
  const float* qkv_w   = (const float*)d_in[3];
  const float* qkv_b   = (const float*)d_in[4];
  const float* proj_w  = (const float*)d_in[5];
  const float* proj_b  = (const float*)d_in[6];
  const float* rpb     = (const float*)d_in[7];
  const float* norm2_w = (const float*)d_in[8];
  const float* norm2_b = (const float*)d_in[9];
  const float* fc1_w   = (const float*)d_in[10];
  const float* fc1_b   = (const float*)d_in[11];
  const float* fc2_w   = (const float*)d_in[12];
  const float* fc2_b   = (const float*)d_in[13];

  char* ws = (char*)d_ws;
  _Float16* wqkv = (_Float16*)(ws + OFF_WQKV);
  _Float16* wprj = (_Float16*)(ws + OFF_WPROJ);
  _Float16* wfc1 = (_Float16*)(ws + OFF_WFC1);
  _Float16* wfc2 = (_Float16*)(ws + OFF_WFC2);
  _Float16* xw   = (_Float16*)(ws + OFF_XW);
  _Float16* qb   = (_Float16*)(ws + OFF_Q);
  _Float16* kb   = (_Float16*)(ws + OFF_K);
  _Float16* vb   = (_Float16*)(ws + OFF_V);
  _Float16* attn = (_Float16*)(ws + OFF_ATTN);
  float*    x1   = (float*)   (ws + OFF_X1);
  _Float16* x2   = xw;
  _Float16* hid  = (_Float16*)(ws + OFF_HID);

  k_cvt_f16<<<(1536 * 512 + 255) / 256, 256, 0, stream>>>(qkv_w, wqkv, 1536 * 512);
  k_cvt_f16<<<(512 * 512 + 255) / 256, 256, 0, stream>>>(proj_w, wprj, 512 * 512);
  k_cvt_f16<<<(2048 * 512 + 255) / 256, 256, 0, stream>>>(fc1_w, wfc1, 2048 * 512);
  k_cvt_f16<<<(512 * 2048 + 255) / 256, 256, 0, stream>>>(fc2_w, wfc2, 512 * 2048);

  hipMemsetAsync(ws + OFF_Q, 0, (size_t)3 * 16777216, stream);

  k_ln_shift_win<<<NT_ / 8, 256, 0, stream>>>(x, norm1_w, norm1_b, xw);
  k_gemm_qkv<<<dim3(6, NT_ / 32), 256, 0, stream>>>(xw, wqkv, qkv_b, qb, kb, vb);
  k_attn<<<NWIN_ * NH_, 32, 0, stream>>>(qb, kb, vb, rpb, attn);
  k_gemm_proj<<<dim3(2, NT_ / 32), 256, 0, stream>>>(attn, wprj, proj_b, x, x1);
  k_ln2<<<NT_ / 8, 256, 0, stream>>>(x1, norm2_w, norm2_b, x2);
  k_gemm_fc1<<<dim3(8, NT_ / 32), 256, 0, stream>>>(x2, wfc1, fc1_b, hid);
  k_gemm_fc2<<<dim3(2, NT_ / 32), 256, 0, stream>>>(hid, wfc2, fc2_b, x1,
                                                    (float*)d_out);
}
